// VQ_54425825574995
// MI455X (gfx1250) — matrix operand order repaired and checked by reading, not compiled, not run on hardware
//
#include <hip/hip_runtime.h>

// Problem constants (from reference)
#define B_     16
#define C_     512
#define T_     4000
#define G_     2
#define D_     256      // features per group (= total WMMA K)
#define K_     320      // codewords
#define NTILES (K_ / 16)
#define TILE_T 256      // tokens per block (8 waves x 32 tokens)
#define PITCH  264      // LDS row pitch in bf16 elems (256 + 8 pad; 16B-aligned rows, bank spread)
#define QOFF   (B_ * C_ * T_)   // offset of indexes in d_out (floats)

typedef __attribute__((ext_vector_type(16))) __bf16 v16bf;
typedef __attribute__((ext_vector_type(8)))  __bf16 v8bf;
typedef __attribute__((ext_vector_type(4)))  __bf16 v4bf;
typedef __attribute__((ext_vector_type(8)))  float  v8f;

__global__ __launch_bounds__(256) void
VQ_54425825574995_kernel(const float* __restrict__ x,
                         const float* __restrict__ cb,
                         float* __restrict__ out)
{
    extern __shared__ char smem[];
    __bf16* A_lds  = (__bf16*)smem;                         // [TILE_T][PITCH] token-major x tile
    __bf16* CB_lds = A_lds + TILE_T * PITCH;                // [K_][PITCH]    bf16 codebook
    float*  bnorm  = (float*)(CB_lds + (size_t)K_ * PITCH); // [K_] ||e_k||^2
    int*    idx_ld = (int*)(bnorm + K_);                    // [TILE_T] argmin result

    const int tid  = threadIdx.x;
    const int lane = tid & 31;
    const int wave = tid >> 5;
    const int t0   = blockIdx.x * TILE_T;
    const int b    = blockIdx.y;
    const int g    = blockIdx.z;

    // ---- Stage codebook fp32 -> bf16 LDS, float4 (b128) global loads ----
    for (int i = tid; i < (K_ * D_) / 4; i += 256) {
        int k  = i >> 6;            // D_/4 = 64 quads per row
        int d4 = (i & 63) * 4;
        float4 v = ((const float4*)cb)[i];
        v4bf p = { (__bf16)v.x, (__bf16)v.y, (__bf16)v.z, (__bf16)v.w };
        *(v4bf*)(CB_lds + k * PITCH + d4) = p;
    }

    // ---- Stage x tile (feature-major -> token-major transpose), float4 loads ----
    // channel of feature d in group g is c = d*2 + g
    const float* xb = x + ((size_t)b * C_ + g) * T_;
    {
        const int tt = (tid & 63) * 4;            // 4 consecutive tokens per thread (256 tokens)
        const int r0 = tid >> 6;                  // 4 feature rows per iteration
        for (int it = 0; it < 64; ++it) {
            int d = it * 4 + r0;
            const float* src = xb + (size_t)d * 2 * T_ + t0 + tt;
            float4 v;
            if (t0 + tt + 3 < T_) {
                v = *(const float4*)src;
            } else {
                v.x = (t0 + tt + 0 < T_) ? src[0] : 0.0f;
                v.y = (t0 + tt + 1 < T_) ? src[1] : 0.0f;
                v.z = (t0 + tt + 2 < T_) ? src[2] : 0.0f;
                v.w = (t0 + tt + 3 < T_) ? src[3] : 0.0f;
            }
            A_lds[(tt + 0) * PITCH + d] = (__bf16)v.x;
            A_lds[(tt + 1) * PITCH + d] = (__bf16)v.y;
            A_lds[(tt + 2) * PITCH + d] = (__bf16)v.z;
            A_lds[(tt + 3) * PITCH + d] = (__bf16)v.w;
        }
    }
    __syncthreads();

    // ---- Codeword squared norms (from the bf16 values actually used) ----
    for (int k = tid; k < K_; k += 256) {
        const __bf16* row = CB_lds + k * PITCH;
        float s = 0.0f;
        for (int d = 0; d < D_; ++d) { float v = (float)row[d]; s += v * v; }
        bnorm[k] = s;
    }
    __syncthreads();

    // ---- Per-wave WMMA GEMM (M=32 per wave: two A-tiles) + fused argmin ----
    const int mb    = wave * 32;            // this wave's 32 tokens
    const int arow  = lane & 15;            // A: lane holds row M = lane%16
    const int koffA = (lane >> 4) * 8;      // A: low lanes K 0..7 & 16..23, high lanes 8..15 & 24..31
    const int ncol  = lane & 15;            // B/C: lane holds column N = lane%16
    const int koffB = (lane >> 4) * 8;      // B: low lanes K 0..7 & 16..23, high lanes 8..15 & 24..31

    v16bf afragA[8], afragB[8];
#pragma unroll
    for (int ks = 0; ks < 8; ++ks) {
        const __bf16* baseA = A_lds + (mb + arow) * PITCH + ks * 32 + koffA;
        const __bf16* baseB = baseA + 16 * PITCH;
        v8bf lo, hi;
        lo = *(const v8bf*)(baseA);
        hi = *(const v8bf*)(baseA + 16);
        afragA[ks] = __builtin_shufflevector(lo, hi, 0,1,2,3,4,5,6,7,8,9,10,11,12,13,14,15);
        lo = *(const v8bf*)(baseB);
        hi = *(const v8bf*)(baseB + 16);
        afragB[ks] = __builtin_shufflevector(lo, hi, 0,1,2,3,4,5,6,7,8,9,10,11,12,13,14,15);
    }

    float bvalA[8], bvalB[8];
    int   bidxA[8], bidxB[8];
#pragma unroll
    for (int r = 0; r < 8; ++r) {
        bvalA[r] = 3.4e38f; bidxA[r] = 0;
        bvalB[r] = 3.4e38f; bidxB[r] = 0;
    }

    // Keep this a real loop (no reassociation blow-up); each iteration:
    // 1 norm load + 16 ds_load_b128 feed 16 WMMAs (two independent acc chains),
    // so the DS-wait cost per WMMA is half that of the M=16 version.
#pragma unroll 1
    for (int nb = 0; nb < NTILES; ++nb) {
        const float nrm = bnorm[nb * 16 + ncol];
        const __bf16* brow = CB_lds + (nb * 16 + ncol) * PITCH + koffB;

        v8f accA = {0.f,0.f,0.f,0.f,0.f,0.f,0.f,0.f};
        v8f accB = {0.f,0.f,0.f,0.f,0.f,0.f,0.f,0.f};
#pragma unroll
        for (int ks = 0; ks < 8; ++ks) {
            v8bf blo = *(const v8bf*)(brow + ks * 32);
            v8bf bhi = *(const v8bf*)(brow + ks * 32 + 16);
            v16bf bfrag = __builtin_shufflevector(blo, bhi,
                          0,1,2,3,4,5,6,7,8,9,10,11,12,13,14,15);
            accA = __builtin_amdgcn_wmma_f32_16x16x32_bf16(
                       false, afragA[ks], false, bfrag, (short)0, accA, false, false);
            accB = __builtin_amdgcn_wmma_f32_16x16x32_bf16(
                       false, afragB[ks], false, bfrag, (short)0, accB, false, false);
        }

        const int cidx = nb * 16 + ncol;
#pragma unroll
        for (int r = 0; r < 8; ++r) {
            float sA = nrm - 2.0f * accA[r];    // ||e||^2 - 2 x.e  (||x||^2 const per token)
            float sB = nrm - 2.0f * accB[r];
            if (sA < bvalA[r]) { bvalA[r] = sA; bidxA[r] = cidx; }
            if (sB < bvalB[r]) { bvalB[r] = sB; bidxB[r] = cidx; }
        }
    }

    // ---- Argmin reduction over the 16 lanes of each half-wave ----
#pragma unroll
    for (int off = 8; off >= 1; off >>= 1) {
#pragma unroll
        for (int r = 0; r < 8; ++r) {
            float ov; int oi;
            ov = __shfl_xor(bvalA[r], off, 32);
            oi = __shfl_xor(bidxA[r], off, 32);
            if (ov < bvalA[r] || (ov == bvalA[r] && oi < bidxA[r])) { bvalA[r] = ov; bidxA[r] = oi; }
            ov = __shfl_xor(bvalB[r], off, 32);
            oi = __shfl_xor(bidxB[r], off, 32);
            if (ov < bvalB[r] || (ov == bvalB[r] && oi < bidxB[r])) { bvalB[r] = ov; bidxB[r] = oi; }
        }
    }
    if ((lane & 15) == 0) {
        int half = lane >> 4;                  // lane0 -> rows 0..7, lane16 -> rows 8..15
#pragma unroll
        for (int r = 0; r < 8; ++r) {
            idx_ld[mb + half * 8 + r]      = bidxA[r];
            idx_ld[mb + 16 + half * 8 + r] = bidxB[r];
        }
    }
    __syncthreads();

    // ---- Epilogue: indexes (as float), layout (G, B, T) ----
    {
        int t = t0 + tid;
        if (t < T_)
            out[QOFF + ((size_t)(g * B_ + b)) * T_ + t] = (float)idx_ld[tid];
    }

    // ---- Epilogue: quantized. One token per thread: fixed codeword row,
    //      float4 row loads (L2-resident), stores coalesced along t. ----
    // output channel for (g, d) is c = g*256 + d (group-concatenated order)
    {
        int t = t0 + tid;
        if (t < T_) {
            int k = idx_ld[tid];
            const float4* crow = (const float4*)(cb + (size_t)k * D_);
            float* orow = out + ((size_t)(b * C_ + g * D_)) * T_ + t;
            for (int d4 = 0; d4 < D_ / 4; ++d4) {
                float4 v = crow[d4];
                orow[(size_t)(d4 * 4 + 0) * T_] = v.x;
                orow[(size_t)(d4 * 4 + 1) * T_] = v.y;
                orow[(size_t)(d4 * 4 + 2) * T_] = v.z;
                orow[(size_t)(d4 * 4 + 3) * T_] = v.w;
            }
        }
    }
}

extern "C" void kernel_launch(void* const* d_in, const int* in_sizes, int n_in,
                              void* d_out, int out_size, void* d_ws, size_t ws_size,
                              hipStream_t stream) {
    (void)in_sizes; (void)n_in; (void)d_ws; (void)ws_size; (void)out_size;
    const float* x  = (const float*)d_in[0];
    const float* cb = (const float*)d_in[1];
    float* out = (float*)d_out;

    size_t smem = (size_t)(TILE_T + K_) * PITCH * sizeof(__bf16)
                + (size_t)K_ * sizeof(float)
                + (size_t)TILE_T * sizeof(int);          // ~300 KB (< 320 KB WGP LDS)
    hipFuncSetAttribute((const void*)VQ_54425825574995_kernel,
                        hipFuncAttributeMaxDynamicSharedMemorySize, (int)smem);

    dim3 grid((T_ + TILE_T - 1) / TILE_T, B_, G_);       // 16 x 16 x 2 = 512 blocks
    VQ_54425825574995_kernel<<<grid, dim3(256), smem, stream>>>(x, cb, out);
}
